// Conv2d_shared_res_bit_baens_61942018343209
// MI455X (gfx1250) — matrix operand order repaired and checked by reading, not compiled, not run on hardware
//
#include <hip/hip_runtime.h>
#include <hip/hip_bf16.h>
#include <math.h>

// ---------------- problem constants ----------------
#define NMEM   6
#define INC    128
#define OUTC   128
#define KSZ    3
#define DTOT   (OUTC*INC*KSZ*KSZ)   // 147456
#define HW     64
#define HWP    66                    // padded 64+2
#define NIMG   48
#define ZP1    1.2f
#define ZP2    (-0.2f)

typedef float  v8f  __attribute__((ext_vector_type(8)));
typedef __bf16 v16bf __attribute__((ext_vector_type(16)));
typedef __bf16 v8bf  __attribute__((ext_vector_type(8)));

union V16U { v16bf v; v8bf h[2]; };

// ---------------- stage 1: min/max of U ----------------
__global__ __launch_bounds__(256) void reduce_minmax_part(
    const float* __restrict__ U, int n, float* __restrict__ pmin, float* __restrict__ pmax)
{
    __shared__ float smin[256];
    __shared__ float smax[256];
    int t = threadIdx.x;
    float mn = 3.4028235e38f, mx = -3.4028235e38f;
    for (int i = blockIdx.x * 256 + t; i < n; i += 256 * gridDim.x) {
        float v = U[i];
        mn = fminf(mn, v);
        mx = fmaxf(mx, v);
    }
    smin[t] = mn; smax[t] = mx;
    __syncthreads();
    for (int s = 128; s > 0; s >>= 1) {
        if (t < s) {
            smin[t] = fminf(smin[t], smin[t + s]);
            smax[t] = fmaxf(smax[t], smax[t + s]);
        }
        __syncthreads();
    }
    if (t == 0) { pmin[blockIdx.x] = smin[0]; pmax[blockIdx.x] = smax[0]; }
}

__global__ __launch_bounds__(256) void reduce_minmax_final(
    const float* __restrict__ pmin, const float* __restrict__ pmax, float* __restrict__ red)
{
    __shared__ float smin[256];
    __shared__ float smax[256];
    int t = threadIdx.x;
    smin[t] = pmin[t]; smax[t] = pmax[t];
    __syncthreads();
    for (int s = 128; s > 0; s >>= 1) {
        if (t < s) {
            smin[t] = fminf(smin[t], smin[t + s]);
            smax[t] = fmaxf(smax[t], smax[t + s]);
        }
        __syncthreads();
    }
    if (t == 0) { red[0] = smin[0]; red[1] = smax[0]; }
}

// ---------------- stage 2: generate bf16 weights in [g][tap][co][ci] ----------------
__global__ __launch_bounds__(256) void gen_weights(
    const float* __restrict__ U, const float* __restrict__ bp, const float* __restrict__ uu,
    const float* __restrict__ red, __bf16* __restrict__ wb)
{
    int d = blockIdx.x * blockDim.x + threadIdx.x;
    if (d >= DTOT) return;

    const float alpha = red[0];
    const float beta  = red[1];
    const float s1 = (beta - alpha) * (1.0f / 3.0f);
    const float s2 = s1 * (1.0f / 5.0f);

    const int kw = d % 3;
    const int kh = (d / 3) % 3;
    const int ci = (d / 9) % INC;
    const int co = d / (INC * 9);
    const int tap = kh * 3 + kw;

    // shared (level-1) gate, row NMEM of bp/u
    {
    }
    float uvS = uu[(size_t)NMEM * DTOT + d];
    float gS  = logf(uvS / (1.0f - uvS));
    float bS  = 1.0f / (1.0f + expf(-(gS + bp[(size_t)NMEM * DTOT + d])));
    float gate1 = fminf(fmaxf(bS * (ZP1 - ZP2) + ZP2, 0.0f), 1.0f);

    float uval[NMEM], gate0[NMEM], v1[NMEM];
    float v1sum = 0.0f;
    #pragma unroll
    for (int n = 0; n < NMEM; ++n) {
        float un = U[(size_t)n * DTOT + d];
        uval[n] = un;
        float uv = uu[(size_t)n * DTOT + d];
        float g  = logf(uv / (1.0f - uv));
        float b  = 1.0f / (1.0f + expf(-(g + bp[(size_t)n * DTOT + d])));
        gate0[n] = fminf(fmaxf(b * (ZP1 - ZP2) + ZP2, 0.0f), 1.0f);
        v1[n] = s1 * rintf(un / s1);        // round half-to-even, matches jnp.round
        v1sum += v1[n];
    }
    const float v1mean = v1sum * (1.0f / (float)NMEM);

    #pragma unroll
    for (int n = 0; n < NMEM; ++n) {
        float v2 = s2 * rintf((uval[n] - v1mean) / s2);
        float og = (gate0[n] > 0.0f) ? 1.0f : 0.0f;     // order gate for level 1
        float w  = v1[n] * gate0[n] + v2 * gate1 * og;
        // layout: [g][tap][co][ci], ci contiguous (WMMA-B friendly)
        wb[(((size_t)n * 9 + tap) * OUTC + co) * INC + ci] = (__bf16)w;
    }
}

// ---------------- stage 3: NCHW f32 -> padded NHWC bf16 ----------------
__global__ __launch_bounds__(256) void pack_nhwc(
    const float* __restrict__ x, __bf16* __restrict__ nhwc)
{
    const int ihp = blockIdx.x;   // 0..65 padded row
    const int img = blockIdx.y;   // 0..47
    const int ih  = ihp - 1;
    __bf16* orow = nhwc + ((size_t)img * HWP + ihp) * HWP * INC;
    const bool rowvalid = (ih >= 0) && (ih < HW);
    const float* xrow = x + (((size_t)img * INC) * HW + (rowvalid ? ih : 0)) * HW;

    // t -> (ci, iwp) with iwp fastest: coalesced reads of x along iw
    for (int t = threadIdx.x; t < INC * HWP; t += blockDim.x) {
        int iwp = t % HWP;
        int ci  = t / HWP;
        int iw  = iwp - 1;
        float v = 0.0f;
        if (rowvalid && iw >= 0 && iw < HW)
            v = xrow[(size_t)ci * HW * HW + iw];
        orow[(size_t)iwp * INC + ci] = (__bf16)v;
    }
}

// ---------------- stage 4: implicit-GEMM conv via bf16 WMMA ----------------
// block: one (image, output row). M = 64 pixels, N = 128 co, K = 9 taps x 128 ci.
// 8 waves: wm = wave&1 selects 32-pixel half, wn = wave>>1 selects 32-co quarter.
// each wave: 2 M-strips x 2 N-strips of 16 => 4 accumulators.
__global__ __launch_bounds__(256) void conv_wmma(
    const __bf16* __restrict__ nhwc, const __bf16* __restrict__ wb,
    float* __restrict__ out)
{
    const int oh   = blockIdx.x;            // 0..63
    const int img  = blockIdx.y;            // 0..47
    const int g    = img % NMEM;
    const int lane = threadIdx.x & 31;
    const int wave = threadIdx.x >> 5;
    const int wm   = wave & 1;
    const int wn   = wave >> 1;
    const int hsel = lane >> 4;             // wave half (0/1)
    const int l16  = lane & 15;

    v8f acc[2][2];
    #pragma unroll
    for (int a = 0; a < 2; ++a)
        #pragma unroll
        for (int b = 0; b < 2; ++b) { v8f z = {}; acc[a][b] = z; }

    const __bf16* wg = wb + (size_t)g * 9 * OUTC * INC;

    for (int kh = 0; kh < 3; ++kh) {
        const __bf16* arow = nhwc + (((size_t)img * HWP + oh + kh) * HWP) * INC;
        for (int kw = 0; kw < 3; ++kw) {
            const __bf16* wtap = wg + (size_t)(kh * 3 + kw) * OUTC * INC;
            #pragma unroll
            for (int kc = 0; kc < 4; ++kc) {
                // A fragments: 16 pixels (M) x 32 ci (K), bf16.
                // lane: m = l16 (pixel), half selects K = h*16 .. h*16+15 (contiguous ci)
                V16U Af[2];
                #pragma unroll
                for (int ms = 0; ms < 2; ++ms) {
                    int ow = wm * 32 + ms * 16 + l16;
                    const __bf16* p = arow + (size_t)(ow + kw) * INC + kc * 32 + hsel * 16;
                    Af[ms].h[0] = *(const v8bf*)(p);
                    Af[ms].h[1] = *(const v8bf*)(p + 8);
                }
                // B fragments: 32 ci (K) x 16 co (N), bf16.
                // lane: n = l16 (co), half selects K = h*16 .. h*16+15 (contiguous ci)
                V16U Bf[2];
                #pragma unroll
                for (int ns = 0; ns < 2; ++ns) {
                    int co = wn * 32 + ns * 16 + l16;
                    const __bf16* p = wtap + (size_t)co * INC + kc * 32 + hsel * 16;
                    Bf[ns].h[0] = *(const v8bf*)(p);
                    Bf[ns].h[1] = *(const v8bf*)(p + 8);
                }
                #pragma unroll
                for (int ms = 0; ms < 2; ++ms)
                    #pragma unroll
                    for (int ns = 0; ns < 2; ++ns)
                        acc[ms][ns] = __builtin_amdgcn_wmma_f32_16x16x32_bf16(
                            false, Af[ms].v, false, Bf[ns].v,
                            (short)0, acc[ms][ns], false, false);
            }
        }
    }

    // D layout: lane column n = l16 (co), VGPR r holds m = r + hsel*8 (pixel).
    // => each lane's 8 floats are 8 consecutive ow pixels: contiguous v8f store.
    #pragma unroll
    for (int ms = 0; ms < 2; ++ms) {
        int owb = wm * 32 + ms * 16 + hsel * 8;
        #pragma unroll
        for (int ns = 0; ns < 2; ++ns) {
            int co = wn * 32 + ns * 16 + l16;
            float* p = out + (((size_t)img * OUTC + co) * HW + oh) * HW + owb;
            *(v8f*)p = acc[ms][ns];
        }
    }
}

// ---------------- launch ----------------
extern "C" void kernel_launch(void* const* d_in, const int* in_sizes, int n_in,
                              void* d_out, int out_size, void* d_ws, size_t ws_size,
                              hipStream_t stream) {
    const float* x  = (const float*)d_in[0];   // (48,128,64,64)
    const float* U  = (const float*)d_in[1];   // (6, D)
    const float* bp = (const float*)d_in[2];   // (7, D)
    const float* uu = (const float*)d_in[3];   // (7, D)
    float* out = (float*)d_out;

    char* ws = (char*)d_ws;
    // [0 .. 1769472)            : bf16 weights [6][9][128][128]
    // [1769472 .. +2048)        : 256 partial mins, 256 partial maxs
    // [+2048 .. +2056)          : red[0]=alpha, red[1]=beta
    // [1773568 .. +53526528)    : padded NHWC bf16 activations [48][66][66][128]
    __bf16* wb   = (__bf16*)ws;
    float*  pmin = (float*)(ws + 1769472);
    float*  pmax = pmin + 256;
    float*  red  = pmax + 256;
    __bf16* nhwc = (__bf16*)(ws + 1769472 + 4096);

    const int nU = NMEM * DTOT;

    reduce_minmax_part<<<256, 256, 0, stream>>>(U, nU, pmin, pmax);
    reduce_minmax_final<<<1, 256, 0, stream>>>(pmin, pmax, red);

    gen_weights<<<(DTOT + 255) / 256, 256, 0, stream>>>(U, bp, uu, red, wb);

    pack_nhwc<<<dim3(HWP, NIMG), 256, 0, stream>>>(x, nhwc);

    conv_wmma<<<dim3(HW, NIMG), 256, 0, stream>>>(nhwc, wb, out);
}
